// CrossAttention_28527172780223
// MI455X (gfx1250) — matrix operand order repaired and checked by reading, not compiled, not run on hardware
//
#include <hip/hip_runtime.h>
#include <hip/hip_bf16.h>

// ---------------------------------------------------------------------------
// Linear-attention block for MI455X (gfx1250, wave32, WMMA).
// All six GEMMs run on v_wmma_f32_16x16x32_bf16 (bf16 operands, f32 accum).
// Softmax / LayerNorm / residual kept in fp32.
// A-fragments are loaded straight from row-major global memory (two b128s per
// lane, matching the ISA A layout) -> no per-K-step LDS staging barriers.
// ---------------------------------------------------------------------------

#define DEV static __device__ __forceinline__

typedef __attribute__((ext_vector_type(16))) __bf16 v16bf;
typedef __attribute__((ext_vector_type(8)))  float  v8f;

constexpr int BATCH = 8;
constexpr int SEQ   = 4096;
constexpr int DIM   = 1024;
constexpr int NROWS = BATCH * SEQ;   // 32768
constexpr int NT    = DIM / 16;      // 64 column tiles of 16

DEV unsigned short f2bf(float f) {
  union { float f; unsigned u; } c; c.f = f;
  const unsigned u = c.u;
  if ((u & 0x7F800000u) == 0x7F800000u) return (unsigned short)(u >> 16); // Inf/NaN
  return (unsigned short)((u + 0x7FFFu + ((u >> 16) & 1u)) >> 16);        // RNE
}
DEV float bf2f(unsigned short h) {
  union { unsigned u; float f; } c; c.u = ((unsigned)h) << 16;
  return c.f;
}

// Load a 16-element bf16 WMMA fragment from two 16-byte chunks.
DEV v16bf load_frag(const unsigned short* p0, const unsigned short* p1) {
  union { v16bf v; float4 q[2]; } u;
  u.q[0] = *reinterpret_cast<const float4*>(p0);
  u.q[1] = *reinterpret_cast<const float4*>(p1);
  return u.v;
}

DEV v8f wmma_bf16(v16bf a, v16bf b, v8f c) {
  // D = A(16x32) * B(32x16) + C(16x16), f32 accumulate
  return __builtin_amdgcn_wmma_f32_16x16x32_bf16(false, a, false, b,
                                                 (short)0, c, false, false);
}

// ---------------------------------------------------------------------------
// Elementwise fp32 -> bf16
// ---------------------------------------------------------------------------
__global__ __launch_bounds__(256) void cvt_f32_bf16_kernel(
    const float* __restrict__ s, unsigned short* __restrict__ d, int n) {
  const int i = blockIdx.x * 256 + threadIdx.x;
  if (i < n) d[i] = f2bf(s[i]);
}

// ---------------------------------------------------------------------------
// Swizzle a DIM x DIM row-major fp32 matrix into the WMMA B-operand layout:
// per (ktile, ntile) 32x16 tile, lane = n + 16*half holds K = 8*half + j (j < 8),
// K = 16 + 8*half + (j - 8) (j >= 8) -> each lane's fragment is one 32-byte run.
// ---------------------------------------------------------------------------
__global__ __launch_bounds__(256) void swz_f32_kernel(
    const float* __restrict__ src, unsigned short* __restrict__ dst) {
  const int idx  = blockIdx.x * 256 + threadIdx.x;   // (tile, lane)
  const int lane = idx & 31, tile = idx >> 5;
  const int nt = tile & (NT - 1), kt = tile >> 6;
  const int hf = lane >> 4, n = lane & 15;
#pragma unroll
  for (int j = 0; j < 16; ++j) {
    const int kk = kt * 32 + 8 * hf + (j & 7) + 16 * (j >> 3);
    dst[(size_t)idx * 16 + j] = f2bf(src[(size_t)kk * DIM + nt * 16 + n]);
  }
}

// ---------------------------------------------------------------------------
// Fused Q/K/V projection.  blockIdx.y selects which projection; the q path
// fuses the feature-dim softmax (* D^-0.5).  One block = 16 rows x 1024 cols,
// 8 waves x 8 WMMA tiles each.  A fragments come straight from global memory
// (row-major == ISA A layout), B fragments from the pre-swizzled weights, so
// the K loop has no barriers at all.
// ---------------------------------------------------------------------------
__global__ __launch_bounds__(256) void qkv_kernel(
    const unsigned short* __restrict__ xb,
    const unsigned short* __restrict__ wqs,
    const unsigned short* __restrict__ wks,
    const unsigned short* __restrict__ wvs,
    unsigned short* __restrict__ qsb,
    unsigned short* __restrict__ kbb,
    unsigned short* __restrict__ vbb) {
  __shared__ float red[16 * 8];
  __shared__ float rowMax[16];
  __shared__ float rowSum[16];

  const int tid  = threadIdx.x;
  const int wave = tid >> 5, lane = tid & 31;
  const int hf = lane >> 4, hl = lane & 15;
  const int row0 = blockIdx.x * 16;
  const int mode = blockIdx.y;   // 0=q 1=k 2=v
  const unsigned short* bsw = (mode == 0) ? wqs : (mode == 1) ? wks : wvs;

  v8f acc[8];
#pragma unroll
  for (int t = 0; t < 8; ++t) acc[t] = {};

  // A fragment: lane m=hl, elems 0..7 = K 8*hf.., elems 8..15 = K 16+8*hf..
  const unsigned short* arow = xb + (size_t)(row0 + hl) * DIM + 8 * hf;

  for (int k0 = 0; k0 < DIM; k0 += 32) {
    const v16bf afrag = load_frag(arow + k0, arow + k0 + 16);
    const int ktile = k0 >> 5;
#pragma unroll
    for (int t = 0; t < 8; ++t) {
      const unsigned short* bp =
          bsw + ((size_t)(ktile * NT + wave * 8 + t) * 32 + lane) * 16;
      if (t == 0) __builtin_prefetch(bp + (size_t)NT * 32 * 16, 0, 3);  // next K step
      const v16bf bfrag = load_frag(bp, bp + 8);
      acc[t] = wmma_bf16(afrag, bfrag, acc[t]);
    }
  }

  if (mode != 0) {   // k / v: raw bf16 store (k softmax handled downstream)
    unsigned short* dst = (mode == 1) ? kbb : vbb;
#pragma unroll
    for (int t = 0; t < 8; ++t) {
      const int col = (wave * 8 + t) * 16 + hl;
#pragma unroll
      for (int r = 0; r < 8; ++r)
        dst[(size_t)(row0 + r + 8 * hf) * DIM + col] = f2bf(acc[t][r]);
    }
    return;
  }

  // q: softmax over the 1024-wide feature dim, then * D^-0.5 (=1/32)
#pragma unroll
  for (int r = 0; r < 8; ++r) {
    float m = -3.0e38f;
#pragma unroll
    for (int t = 0; t < 8; ++t) m = fmaxf(m, acc[t][r]);
#pragma unroll
    for (int off = 8; off > 0; off >>= 1) m = fmaxf(m, __shfl_xor(m, off, 16));
    if (hl == 0) red[(r + 8 * hf) * 8 + wave] = m;
  }
  __syncthreads();
  if (tid < 16) {
    float m = -3.0e38f;
    for (int w = 0; w < 8; ++w) m = fmaxf(m, red[tid * 8 + w]);
    rowMax[tid] = m;
  }
  __syncthreads();
#pragma unroll
  for (int r = 0; r < 8; ++r) {
    const float rm = rowMax[r + 8 * hf];
    float s = 0.f;
#pragma unroll
    for (int t = 0; t < 8; ++t) s += __expf(acc[t][r] - rm);
#pragma unroll
    for (int off = 8; off > 0; off >>= 1) s += __shfl_xor(s, off, 16);
    if (hl == 0) red[(r + 8 * hf) * 8 + wave] = s;
  }
  __syncthreads();
  if (tid < 16) {
    float s = 0.f;
    for (int w = 0; w < 8; ++w) s += red[tid * 8 + w];
    rowSum[tid] = s;
  }
  __syncthreads();
  const float qscale = 0.03125f;   // 1024^-0.5
#pragma unroll
  for (int t = 0; t < 8; ++t) {
    const int col = (wave * 8 + t) * 16 + hl;
#pragma unroll
    for (int r = 0; r < 8; ++r) {
      const int m = r + 8 * hf;
      const float v = __expf(acc[t][r] - rowMax[m]) / rowSum[m] * qscale;
      qsb[(size_t)(row0 + m) * DIM + col] = f2bf(v);
    }
  }
}

// ---------------------------------------------------------------------------
// Sequence-dim softmax stats for k: per (b,d) online max & sum(exp) over L.
// Consecutive threads = consecutive d -> coalesced rows.
// ---------------------------------------------------------------------------
__global__ __launch_bounds__(256) void kstats_kernel(
    const unsigned short* __restrict__ kbb,
    float* __restrict__ kmax, float* __restrict__ ksum) {
  const int col = blockIdx.x * 256 + threadIdx.x;   // [0, BATCH*DIM)
  const int b = col >> 10, d = col & (DIM - 1);
  const unsigned short* kp = kbb + (size_t)b * SEQ * DIM + d;
  float m = -3.0e38f, s = 0.f;
  for (int l = 0; l < SEQ; ++l) {
    const float x = bf2f(kp[(size_t)l * DIM]);
    const float nm = fmaxf(m, x);
    s = s * __expf(m - nm) + __expf(x - nm);
    m = nm;
  }
  kmax[col] = m;
  ksum[col] = s;
}

// ---------------------------------------------------------------------------
// context[b] = softmax_seq(k)^T @ v   (1024x1024 per batch, K = L = 4096).
// k-normalization fused into the LDS A-stage; v transposed through LDS so
// each lane's B fragment is two 16B runs, 32B apart (two ds_load_b128).
// Wave owns four 16x16 output tiles.  The result is written DIRECTLY in the
// WMMA B-operand swizzled layout consumed by the attention GEMM (accumulator
// rows map to contiguous j indices -> one packed b128 store per tile).
// ---------------------------------------------------------------------------
__global__ __launch_bounds__(256) void ctx_kernel(
    const unsigned short* __restrict__ kbb,
    const unsigned short* __restrict__ vbb,
    const float* __restrict__ kmax, const float* __restrict__ ksum,
    unsigned short* __restrict__ ctxs) {
  __shared__ __align__(16) unsigned short lA[16 * 32];
  __shared__ __align__(16) unsigned short lBt[8][4][16 * 32];   // 32 KB

  const int tid  = threadIdx.x;
  const int wave = tid >> 5, lane = tid & 31;
  const int hf = lane >> 4, hl = lane & 15;
  const int b   = blockIdx.x >> 6;
  const int dt  = blockIdx.x & 63;               // 16-wide d tile
  const int d0  = dt * 16;
  const int et0 = blockIdx.y * 32 + wave * 4;    // first of 4 e tiles
  const int e0  = et0 * 16;

  v8f acc[4];
#pragma unroll
  for (int t = 0; t < 4; ++t) acc[t] = {};

  for (int l0 = 0; l0 < SEQ; l0 += 32) {
    __syncthreads();
    // A[m][kk] = exp(k[l0+kk][d0+m] - max_d) / sum_d   (k^T with softmax fused)
    for (int e = tid; e < 512; e += 256) {
      const int kk = e >> 4, m = e & 15;
      const int d = d0 + m;
      const float kv = bf2f(kbb[((size_t)b * SEQ + l0 + kk) * DIM + d]);
      lA[m * 32 + kk] = f2bf(__expf(kv - kmax[b * DIM + d]) / ksum[b * DIM + d]);
    }
    // B tiles of v, transposed: lBt[n][kk] = v[l0+kk][e0+n]
#pragma unroll
    for (int t = 0; t < 4; ++t) {
      const unsigned short* vp = vbb + ((size_t)b * SEQ + l0 + lane) * DIM + e0 + t * 16;
      union { uint4 q[2]; unsigned short s[16]; } tmp;
      tmp.q[0] = reinterpret_cast<const uint4*>(vp)[0];
      tmp.q[1] = reinterpret_cast<const uint4*>(vp)[1];
#pragma unroll
      for (int n = 0; n < 16; ++n) lBt[wave][t][n * 32 + lane] = tmp.s[n];
    }
    __syncthreads();
    const unsigned short* ar = lA + hl * 32 + 8 * hf;
    const v16bf afrag = load_frag(ar, ar + 16);
#pragma unroll
    for (int t = 0; t < 4; ++t) {
      const unsigned short* br = &lBt[wave][t][hl * 32 + 8 * hf];
      const v16bf bfrag = load_frag(br, br + 16);
      acc[t] = wmma_bf16(afrag, bfrag, acc[t]);
    }
  }

  // direct store into B-swizzled layout: tile (kt = d/32, nt = e/16),
  // lane_s = (e&15) + 16*hf, j = 8*((d>>4)&1) + r  (contiguous in r).
  unsigned short* cb = ctxs + (size_t)b * DIM * DIM;
  const int kt = dt >> 1, hfs = dt & 1;
#pragma unroll
  for (int t = 0; t < 4; ++t) {
    union { uint4 q; unsigned short s[8]; } o;
#pragma unroll
    for (int r = 0; r < 8; ++r) o.s[r] = f2bf(acc[t][r]);
    const size_t base =
        ((size_t)(kt * NT + et0 + t) * 32 + hl + 16 * hf) * 16 + 8 * hfs;
    *reinterpret_cast<uint4*>(cb + base) = o.q;
  }
}

// ---------------------------------------------------------------------------
// Fused: attn = q_soft @ ctx[b]  -> (bf16, kept in LDS)
//        out  = attn @ Wo + bo + x  -> LayerNorm -> fp32
// Phase 1 A fragments come straight from global (no barriers in K loop);
// phase 2 A fragments are read straight from LDS attn tile.
// ---------------------------------------------------------------------------
__global__ __launch_bounds__(256) void attn_out_ln_kernel(
    const unsigned short* __restrict__ qsb,
    const unsigned short* __restrict__ ctxs,
    const unsigned short* __restrict__ wos,
    const float* __restrict__ x,
    const float* __restrict__ bo,
    const float* __restrict__ gamma,
    const float* __restrict__ beta,
    float* __restrict__ out) {
  __shared__ __align__(16) unsigned short lAttn[16 * DIM];   // 32 KB
  __shared__ float red[16 * 8];
  __shared__ float red2[16 * 8];
  __shared__ float rMu[16];
  __shared__ float rRs[16];

  const int tid  = threadIdx.x;
  const int wave = tid >> 5, lane = tid & 31;
  const int hf = lane >> 4, hl = lane & 15;
  const int row0 = blockIdx.x * 16;
  const int b = row0 >> 12;   // row0 / SEQ
  const unsigned short* cs = ctxs + (size_t)b * DIM * DIM;

  v8f acc[8];
#pragma unroll
  for (int t = 0; t < 8; ++t) acc[t] = {};

  // phase 1: attn = q_soft @ context
  {
    const unsigned short* arow = qsb + (size_t)(row0 + hl) * DIM + 8 * hf;
    for (int k0 = 0; k0 < DIM; k0 += 32) {
      const v16bf afrag = load_frag(arow + k0, arow + k0 + 16);
      const int ktile = k0 >> 5;
#pragma unroll
      for (int t = 0; t < 8; ++t) {
        const unsigned short* bp =
            cs + ((size_t)(ktile * NT + wave * 8 + t) * 32 + lane) * 16;
        if (t == 0) __builtin_prefetch(bp + (size_t)NT * 32 * 16, 0, 3);
        const v16bf bfrag = load_frag(bp, bp + 8);
        acc[t] = wmma_bf16(afrag, bfrag, acc[t]);
      }
    }
  }
#pragma unroll
  for (int t = 0; t < 8; ++t) {
    const int col = (wave * 8 + t) * 16 + hl;
#pragma unroll
    for (int r = 0; r < 8; ++r)
      lAttn[(r + 8 * hf) * DIM + col] = f2bf(acc[t][r]);
  }
  __syncthreads();

  // phase 2: out = attn @ Wo  (A fragments read straight from LDS)
#pragma unroll
  for (int t = 0; t < 8; ++t) acc[t] = {};
  for (int k0 = 0; k0 < DIM; k0 += 32) {
    const unsigned short* ar = lAttn + hl * DIM + k0 + 8 * hf;
    const v16bf afrag = load_frag(ar, ar + 16);
    const int ktile = k0 >> 5;
#pragma unroll
    for (int t = 0; t < 8; ++t) {
      const unsigned short* bp =
          wos + ((size_t)(ktile * NT + wave * 8 + t) * 32 + lane) * 16;
      if (t == 0) __builtin_prefetch(bp + (size_t)NT * 32 * 16, 0, 3);
      const v16bf bfrag = load_frag(bp, bp + 8);
      acc[t] = wmma_bf16(afrag, bfrag, acc[t]);
    }
  }

  // epilogue: + bo + residual, LayerNorm over the 1024-wide row
#pragma unroll
  for (int t = 0; t < 8; ++t) {
    const int col = (wave * 8 + t) * 16 + hl;
    const float bv = bo[col];
#pragma unroll
    for (int r = 0; r < 8; ++r)
      acc[t][r] = acc[t][r] + bv + x[(size_t)(row0 + r + 8 * hf) * DIM + col];
  }
#pragma unroll
  for (int r = 0; r < 8; ++r) {
    float s = 0.f, q = 0.f;
#pragma unroll
    for (int t = 0; t < 8; ++t) { const float v = acc[t][r]; s += v; q += v * v; }
#pragma unroll
    for (int off = 8; off > 0; off >>= 1) {
      s += __shfl_xor(s, off, 16);
      q += __shfl_xor(q, off, 16);
    }
    if (hl == 0) { red[(r + 8 * hf) * 8 + wave] = s; red2[(r + 8 * hf) * 8 + wave] = q; }
  }
  __syncthreads();
  if (tid < 16) {
    float s = 0.f, q = 0.f;
    for (int w = 0; w < 8; ++w) { s += red[tid * 8 + w]; q += red2[tid * 8 + w]; }
    const float mu = s * (1.f / DIM);
    rMu[tid] = mu;
    rRs[tid] = rsqrtf(q * (1.f / DIM) - mu * mu + 1e-5f);
  }
  __syncthreads();
#pragma unroll
  for (int t = 0; t < 8; ++t) {
    const int col = (wave * 8 + t) * 16 + hl;
    const float g = gamma[col], be = beta[col];
#pragma unroll
    for (int r = 0; r < 8; ++r) {
      const int m = r + 8 * hf;
      out[(size_t)(row0 + m) * DIM + col] = (acc[t][r] - rMu[m]) * rRs[m] * g + be;
    }
  }
}

// ---------------------------------------------------------------------------
extern "C" void kernel_launch(void* const* d_in, const int* in_sizes, int n_in,
                              void* d_out, int out_size, void* d_ws, size_t ws_size,
                              hipStream_t stream) {
  (void)in_sizes; (void)n_in; (void)out_size; (void)ws_size;
  const float* x     = (const float*)d_in[0];
  const float* Wq    = (const float*)d_in[1];
  const float* Wk    = (const float*)d_in[2];
  const float* Wv    = (const float*)d_in[3];
  const float* Wo    = (const float*)d_in[4];
  const float* bo    = (const float*)d_in[5];
  const float* gamma = (const float*)d_in[6];
  const float* beta  = (const float*)d_in[7];
  float* out = (float*)d_out;

  char* p = (char*)d_ws;
  auto take = [&p](size_t bytes) {
    char* r = p;
    p += (bytes + 255) & ~(size_t)255;
    return r;
  };
  unsigned short* xb   = (unsigned short*)take((size_t)NROWS * DIM * 2);      // x bf16
  unsigned short* wqs  = (unsigned short*)take((size_t)DIM * DIM * 2);        // swizzled
  unsigned short* wks  = (unsigned short*)take((size_t)DIM * DIM * 2);
  unsigned short* wvs  = (unsigned short*)take((size_t)DIM * DIM * 2);
  unsigned short* wos  = (unsigned short*)take((size_t)DIM * DIM * 2);
  unsigned short* qsb  = (unsigned short*)take((size_t)NROWS * DIM * 2);      // softmaxed q
  unsigned short* kbb  = (unsigned short*)take((size_t)NROWS * DIM * 2);      // raw k
  unsigned short* vbb  = (unsigned short*)take((size_t)NROWS * DIM * 2);      // v
  unsigned short* ctxs = (unsigned short*)take((size_t)BATCH * DIM * DIM * 2);// swizzled ctx
  float* kmax = (float*)take((size_t)BATCH * DIM * 4);
  float* ksum = (float*)take((size_t)BATCH * DIM * 4);

  const int nX = NROWS * DIM;
  cvt_f32_bf16_kernel<<<nX / 256, 256, 0, stream>>>(x, xb, nX);
  swz_f32_kernel<<<256, 256, 0, stream>>>(Wq, wqs);
  swz_f32_kernel<<<256, 256, 0, stream>>>(Wk, wks);
  swz_f32_kernel<<<256, 256, 0, stream>>>(Wv, wvs);
  swz_f32_kernel<<<256, 256, 0, stream>>>(Wo, wos);
  qkv_kernel<<<dim3(NROWS / 16, 3), 256, 0, stream>>>(xb, wqs, wks, wvs, qsb, kbb, vbb);
  kstats_kernel<<<(BATCH * DIM) / 256, 256, 0, stream>>>(kbb, kmax, ksum);
  ctx_kernel<<<dim3(BATCH * (DIM / 16), DIM / 512), 256, 0, stream>>>(kbb, vbb, kmax, ksum, ctxs);
  attn_out_ln_kernel<<<NROWS / 16, 256, 0, stream>>>(qsb, ctxs, wos, x, bo, gamma, beta, out);
}
